// MultiHeadAttention_24653112279557
// MI455X (gfx1250) — matrix operand order repaired and checked by reading, not compiled, not run on hardware
//
#include <hip/hip_runtime.h>
#include <hip/hip_bf16.h>
#include <math.h>

#define S_LEN   2048
#define DMODEL  1024
#define HEADS   16
#define DK      64

typedef unsigned short u16;
typedef __attribute__((ext_vector_type(16))) __bf16 v16bf;
typedef __attribute__((ext_vector_type(8)))  float  v8f;

union FragU { v16bf v; uint4 q[2]; };

__device__ __forceinline__ u16 f2bf(float f) {
  unsigned u = __float_as_uint(f);
  u += 0x7FFFu + ((u >> 16) & 1u);          // round-to-nearest-even
  return (u16)(u >> 16);
}

__device__ __forceinline__ v8f v8f_zero() {
  v8f z;
#pragma unroll
  for (int i = 0; i < 8; ++i) z[i] = 0.0f;
  return z;
}

__device__ __forceinline__ v8f wmma_bf16(v16bf a, v16bf b, v8f c) {
  return __builtin_amdgcn_wmma_f32_16x16x32_bf16(false, a, false, b,
                                                 (short)0, c, false, false);
}

// A fragment 16x32 bf16 (ISA 7.12.2): lane m = l&15.
// lanes 0-15 hold K 0..7 & 16..23, lanes 16-31 hold K 8..15 & 24..31.
__device__ __forceinline__ v16bf load_a_frag(const u16* __restrict__ base, int ld,
                                             int row, int k0, int lane) {
  const u16* p = base + (size_t)row * ld + k0;
  const int hs = (lane >> 4) * 8;
  FragU f;
  f.q[0] = *(const uint4*)(p + hs);
  f.q[1] = *(const uint4*)(p + 16 + hs);
  return f.v;
}

// B fragment 32x16 bf16 from a row-major source whose row n holds the K dim
// contiguously.  lane n = l&15; lanes 0-15 hold K 0..7 & 16..23, lanes 16-31 K 8..15 & 24..31.
__device__ __forceinline__ v16bf load_b_frag(const u16* __restrict__ base, int ld,
                                             int nrow, int k0, int lane) {
  const u16* p = base + (size_t)nrow * ld + k0 + (lane >> 4) * 8;
  FragU f;
  f.q[0] = *(const uint4*)(p);
  f.q[1] = *(const uint4*)(p + 16);
  return f.v;
}

// B fragment from LDS (row-major [nrows][ld] halfs).
__device__ __forceinline__ v16bf load_b_frag_lds(const u16* base, int ld,
                                                 int nrow, int k0, int lane) {
  const u16* p = base + nrow * ld + k0 + (lane >> 4) * 8;
  FragU f;
  f.q[0] = *(const uint4*)(p);
  f.q[1] = *(const uint4*)(p + 16);
  return f.v;
}

// ---- CDNA5 async global->LDS copy (ASYNCcnt-tracked), per-lane 16 bytes.
__device__ __forceinline__ void async_load_b128(unsigned lds_addr,
                                                const u16* gaddr) {
  asm volatile("global_load_async_to_lds_b128 %0, %1, off"
               :: "v"(lds_addr), "v"(gaddr) : "memory");
}
__device__ __forceinline__ void wait_async_le4() {
  asm volatile("s_wait_asynccnt 0x4" ::: "memory");
}
__device__ __forceinline__ void wait_async_0() {
  asm volatile("s_wait_asynccnt 0x0" ::: "memory");
}
__device__ __forceinline__ void wait_ds_0() {
  asm volatile("s_wait_dscnt 0x0" ::: "memory");
}

// ---------------------------------------------------------------- stage 1
__global__ __launch_bounds__(256)
void cvt_bf16_kernel(const float* __restrict__ in, u16* __restrict__ out, int n4) {
  int i = blockIdx.x * 256 + threadIdx.x;
  if (i >= n4) return;
  float4 f = ((const float4*)in)[i];
  uint2 u;
  u.x = (unsigned)f2bf(f.x) | ((unsigned)f2bf(f.y) << 16);
  u.y = (unsigned)f2bf(f.z) | ((unsigned)f2bf(f.w) << 16);
  ((uint2*)out)[i] = u;
}

// ---------------------------------------------------------------- stage 2
// C = x * W^T (M=8192, N=1024, K=1024), wave tile 64x32, block tile 128x128.
// z = 0/1/2 -> Q/K/V.  Q,K: RoPE in f32 epilogue, stored [B,H,S,64] bf16.
// V stored transposed [B,H,64,S] bf16 so PV B-frags are contiguous.
__global__ __launch_bounds__(256)
void qkv_rope_kernel(const u16* __restrict__ xb,  const u16* __restrict__ wqb,
                     const u16* __restrict__ wkb, const u16* __restrict__ wvb,
                     u16* __restrict__ qmat, u16* __restrict__ kmat,
                     u16* __restrict__ vtb) {
  const int lane  = threadIdx.x & 31;
  const int wave  = threadIdx.x >> 5;
  const int Mbase = blockIdx.x * 128 + (wave >> 2) * 64;
  const int Nbase = blockIdx.y * 128 + (wave & 3) * 32;
  const int which = blockIdx.z;
  const u16* w = (which == 0) ? wqb : (which == 1) ? wkb : wvb;

  v8f acc[4][2];
#pragma unroll
  for (int mt = 0; mt < 4; ++mt)
#pragma unroll
    for (int nt = 0; nt < 2; ++nt) acc[mt][nt] = v8f_zero();

  for (int k0 = 0; k0 < DMODEL; k0 += 32) {
    v16bf a[4], b[2];
#pragma unroll
    for (int mt = 0; mt < 4; ++mt)
      a[mt] = load_a_frag(xb, DMODEL, Mbase + mt * 16 + (lane & 15), k0, lane);
#pragma unroll
    for (int nt = 0; nt < 2; ++nt)
      b[nt] = load_b_frag(w, DMODEL, Nbase + nt * 16 + (lane & 15), k0, lane);
#pragma unroll
    for (int mt = 0; mt < 4; ++mt)
#pragma unroll
      for (int nt = 0; nt < 2; ++nt)
        acc[mt][nt] = wmma_bf16(a[mt], b[nt], acc[mt][nt]);
  }

  const int radd  = (lane >> 4) << 3;
  const int nlane = lane & 15;
#pragma unroll
  for (int mt = 0; mt < 4; ++mt)
#pragma unroll
    for (int nt = 0; nt < 2; ++nt)
#pragma unroll
      for (int r = 0; r < 8; ++r) {
        const int row = Mbase + mt * 16 + r + radd;     // b*2048 + s
        const int col = Nbase + nt * 16 + nlane;        // h*64 + d
        const int bb = row >> 11, s = row & (S_LEN - 1);
        const int h  = col >> 6,  d = col & (DK - 1);
        float v = acc[mt][nt][r];
        if (which < 2) {
          // RoPE: partner feature (d^1) lives in lane^1, same row/VGPR.
          float prt = __shfl_xor(v, 1, 32);
          float inv_freq = __expf(-0.14391156f * (float)(d >> 1)); // ln(1e4)/64
          float ang = (float)s * inv_freq;
          float sn, cs;
          __sincosf(ang, &sn, &cs);
          v = v * cs + ((d & 1) ? prt : -prt) * sn;
          u16* dst = (which == 0) ? qmat : kmat;
          dst[(((size_t)(bb * HEADS + h)) * S_LEN + s) * DK + d] = f2bf(v);
        } else {
          vtb[(((size_t)(bb * HEADS + h)) * DK + d) * S_LEN + s] = f2bf(v);
        }
      }
}

// ---------------------------------------------------------------- stage 3
// Flash attention, causal.  4 waves / block, 128 q rows / block.
// K/V chunks (32 key positions) are double-buffered in LDS via the CDNA5
// async global->LDS path and shared by all 4 waves (4x less L2 traffic).
__global__ __launch_bounds__(128)
void flash_attn_kernel(const u16* __restrict__ qmat, const u16* __restrict__ kmat,
                       const u16* __restrict__ vtb, u16* __restrict__ aout) {
  const int tid  = threadIdx.x;
  const int lane = tid & 31;
  const int wave = tid >> 5;
  const int qblk = blockIdx.x;               // 128 q rows per block
  const int bh   = blockIdx.y;
  const int bb = bh >> 4, h = bh & (HEADS - 1);
  const int q0 = qblk * 128 + wave * 32;     // this wave's 32-row q tile
  const u16* Q = qmat + (size_t)bh * S_LEN * DK;
  const u16* K = kmat + (size_t)bh * S_LEN * DK;
  const u16* V = vtb  + (size_t)bh * DK * S_LEN;

  __shared__ u16 Kbuf[2][32 * DK];    // [chunk rows][d]          4 KB x2
  __shared__ u16 Vbuf[2][DK * 32];    // [d][chunk rows]          4 KB x2
  __shared__ u16 pbuf[4][32 * 32];    // per-wave P bounce        8 KB

  const int radd  = (lane >> 4) << 3;
  const int nlane = lane & 15;

  // Q fragments resident for the whole kernel.
  v16bf qf[2][2];
#pragma unroll
  for (int mt = 0; mt < 2; ++mt)
#pragma unroll
    for (int ks = 0; ks < 2; ++ks)
      qf[mt][ks] = load_a_frag(Q, DK, q0 + mt * 16 + nlane, ks * 32, lane);

  v8f o[2][4];
  float mrow[2][8], lrow[2][8];
#pragma unroll
  for (int mt = 0; mt < 2; ++mt) {
#pragma unroll
    for (int nd = 0; nd < 4; ++nd) o[mt][nd] = v8f_zero();
#pragma unroll
    for (int r = 0; r < 8; ++r) { mrow[mt][r] = -3.0e38f; lrow[mt][r] = 0.f; }
  }

  // Issue async copies of chunk c into buffer buf (4 b128 per thread).
  auto issue = [&](int c, int buf) {
    // K chunk: 32 rows x 64 halfs, fully contiguous 4 KB.
    const u16* kg = K + (size_t)c * 32 * DK;
    unsigned kl = (unsigned)(size_t)&Kbuf[buf][0];
    async_load_b128(kl + tid * 16,        kg + tid * 8);
    async_load_b128(kl + tid * 16 + 2048, kg + tid * 8 + 1024);
    // V chunk: 64 rows (d) x 32 halfs from V^T, 64 B per row.
    const int vrow = tid >> 1, vh = (tid & 1) * 16;
    const u16* vg = V + (size_t)vrow * S_LEN + c * 32 + vh;
    unsigned vl = (unsigned)(size_t)&Vbuf[buf][0] + (vrow * 32 + vh) * 2;
    async_load_b128(vl,      vg);
    async_load_b128(vl + 16, vg + 8);
  };

  const int nchunks = qblk * 4 + 4;          // covers kc < qblk*128+128
  issue(0, 0);

  for (int c = 0; c < nchunks; ++c) {
    if (c + 1 < nchunks) { issue(c + 1, (c + 1) & 1); wait_async_le4(); }
    else                 { wait_async_0(); }
    __syncthreads();                          // chunk c visible to all waves

    const int kc = c * 32;
    if (kc < q0 + 32) {                       // this wave still has work
      const u16* kb = &Kbuf[c & 1][0];
      const u16* vb = &Vbuf[c & 1][0];

      // ---- S = Q K^T (f32 accum), k-dim 64 -> 2 WMMA steps
      v8f sacc[2][2];
      v16bf kf[2][2];
#pragma unroll
      for (int mt = 0; mt < 2; ++mt)
#pragma unroll
        for (int nt = 0; nt < 2; ++nt) sacc[mt][nt] = v8f_zero();
#pragma unroll
      for (int nt = 0; nt < 2; ++nt)
#pragma unroll
        for (int ks = 0; ks < 2; ++ks)
          kf[nt][ks] = load_b_frag_lds(kb, DK, nt * 16 + nlane, ks * 32, lane);
#pragma unroll
      for (int mt = 0; mt < 2; ++mt)
#pragma unroll
        for (int nt = 0; nt < 2; ++nt)
#pragma unroll
          for (int ks = 0; ks < 2; ++ks)
            sacc[mt][nt] = wmma_bf16(qf[mt][ks], kf[nt][ks], sacc[mt][nt]);

      const bool diag = (kc == q0);

      // ---- online softmax (row stats lane-local in the C layout)
#pragma unroll
      for (int mt = 0; mt < 2; ++mt) {
        float sv[2][8];
#pragma unroll
        for (int nt = 0; nt < 2; ++nt)
#pragma unroll
          for (int r = 0; r < 8; ++r) {
            float x = sacc[mt][nt][r] * 0.125f;     // 1/sqrt(64)
            if (diag) {
              int qpos = q0 + mt * 16 + r + radd;
              int kpos = kc + nt * 16 + nlane;
              if (kpos > qpos) x = -3.0e38f;
            }
            sv[nt][r] = x;
          }
#pragma unroll
        for (int r = 0; r < 8; ++r) {
          float vm = fmaxf(sv[0][r], sv[1][r]);
          vm = fmaxf(vm, __shfl_xor(vm, 1, 32));
          vm = fmaxf(vm, __shfl_xor(vm, 2, 32));
          vm = fmaxf(vm, __shfl_xor(vm, 4, 32));
          vm = fmaxf(vm, __shfl_xor(vm, 8, 32));
          float mold = mrow[mt][r];
          float mnew = fmaxf(mold, vm);
          float alpha = __expf(mold - mnew);
          mrow[mt][r] = mnew;
          float p0 = __expf(sv[0][r] - mnew);
          float p1 = __expf(sv[1][r] - mnew);
          float rs = p0 + p1;
          rs += __shfl_xor(rs, 1, 32);
          rs += __shfl_xor(rs, 2, 32);
          rs += __shfl_xor(rs, 4, 32);
          rs += __shfl_xor(rs, 8, 32);
          lrow[mt][r] = lrow[mt][r] * alpha + rs;
#pragma unroll
          for (int nd = 0; nd < 4; ++nd) o[mt][nd][r] *= alpha;
          const int prow = mt * 16 + r + radd;
          pbuf[wave][prow * 32 + nlane]      = f2bf(p0);
          pbuf[wave][prow * 32 + 16 + nlane] = f2bf(p1);
        }
      }
      // Wave-private bounce: DS ops from one wave are in-order (ISA 5.7);
      // the wait + memory clobber just pins the store->load order.
      wait_ds_0();

      // ---- O += P V
      v16bf pf[2], vf[4];
#pragma unroll
      for (int mt = 0; mt < 2; ++mt) {
        const u16* p = &pbuf[wave][(mt * 16 + nlane) * 32];
        FragU f;
        f.q[0] = *(const uint4*)(p + (lane >> 4) * 8);
        f.q[1] = *(const uint4*)(p + 16 + (lane >> 4) * 8);
        pf[mt] = f.v;
      }
#pragma unroll
      for (int nd = 0; nd < 4; ++nd)
        vf[nd] = load_b_frag_lds(vb, 32, nd * 16 + nlane, 0, lane);
#pragma unroll
      for (int mt = 0; mt < 2; ++mt)
#pragma unroll
        for (int nd = 0; nd < 4; ++nd)
          o[mt][nd] = wmma_bf16(pf[mt], vf[nd], o[mt][nd]);
    }
    __syncthreads();        // all reads of buffer (c&1) done before reuse
  }

  // ---- normalize and store in [B,S,H*dk] layout (ready for W_O GEMM)
#pragma unroll
  for (int mt = 0; mt < 2; ++mt)
#pragma unroll
    for (int nd = 0; nd < 4; ++nd)
#pragma unroll
      for (int r = 0; r < 8; ++r) {
        float ov = o[mt][nd][r] / lrow[mt][r];
        int s = q0 + mt * 16 + r + radd;
        int d = nd * 16 + nlane;
        aout[((size_t)(bb * S_LEN + s)) * DMODEL + h * DK + d] = f2bf(ov);
      }
}

// ---------------------------------------------------------------- stage 4
__global__ __launch_bounds__(256)
void out_proj_kernel(const u16* __restrict__ ab, const u16* __restrict__ wob,
                     float* __restrict__ out) {
  const int lane  = threadIdx.x & 31;
  const int wave  = threadIdx.x >> 5;
  const int Mbase = blockIdx.x * 128 + (wave >> 2) * 64;
  const int Nbase = blockIdx.y * 128 + (wave & 3) * 32;

  v8f acc[4][2];
#pragma unroll
  for (int mt = 0; mt < 4; ++mt)
#pragma unroll
    for (int nt = 0; nt < 2; ++nt) acc[mt][nt] = v8f_zero();

  for (int k0 = 0; k0 < DMODEL; k0 += 32) {
    v16bf a[4], b[2];
#pragma unroll
    for (int mt = 0; mt < 4; ++mt)
      a[mt] = load_a_frag(ab, DMODEL, Mbase + mt * 16 + (lane & 15), k0, lane);
#pragma unroll
    for (int nt = 0; nt < 2; ++nt)
      b[nt] = load_b_frag(wob, DMODEL, Nbase + nt * 16 + (lane & 15), k0, lane);
#pragma unroll
    for (int mt = 0; mt < 4; ++mt)
#pragma unroll
      for (int nt = 0; nt < 2; ++nt)
        acc[mt][nt] = wmma_bf16(a[mt], b[nt], acc[mt][nt]);
  }

  const int radd  = (lane >> 4) << 3;
  const int nlane = lane & 15;
#pragma unroll
  for (int mt = 0; mt < 4; ++mt)
#pragma unroll
    for (int nt = 0; nt < 2; ++nt)
#pragma unroll
      for (int r = 0; r < 8; ++r) {
        const int row = Mbase + mt * 16 + r + radd;
        const int col = Nbase + nt * 16 + nlane;
        out[(size_t)row * DMODEL + col] = acc[mt][nt][r];
      }
}

// ----------------------------------------------------------------
extern "C" void kernel_launch(void* const* d_in, const int* in_sizes, int n_in,
                              void* d_out, int out_size, void* d_ws, size_t ws_size,
                              hipStream_t stream) {
  const float* x  = (const float*)d_in[0];
  // d_in[1] = token_positions (int32, arange(S) per batch) -> computed inline
  const float* wq = (const float*)d_in[2];
  const float* wk = (const float*)d_in[3];
  const float* wv = (const float*)d_in[4];
  const float* wo = (const float*)d_in[5];
  (void)in_sizes; (void)n_in; (void)out_size; (void)ws_size;

  const size_t NX = (size_t)4 * S_LEN * DMODEL;   // 8,388,608
  const size_t NW = (size_t)DMODEL * DMODEL;      // 1,048,576

  char* ws = (char*)d_ws;
  u16* xb  = (u16*)ws; ws += NX * 2;
  u16* wqb = (u16*)ws; ws += NW * 2;
  u16* wkb = (u16*)ws; ws += NW * 2;
  u16* wvb = (u16*)ws; ws += NW * 2;
  u16* wob = (u16*)ws; ws += NW * 2;
  u16* qm  = (u16*)ws; ws += NX * 2;
  u16* km  = (u16*)ws; ws += NX * 2;
  u16* vt  = (u16*)ws; ws += NX * 2;
  u16* ao  = (u16*)ws; ws += NX * 2;   // total 88 MB scratch

  cvt_bf16_kernel<<<(int)(NX / 4 / 256), 256, 0, stream>>>(x,  xb,  (int)(NX / 4));
  cvt_bf16_kernel<<<(int)(NW / 4 / 256), 256, 0, stream>>>(wq, wqb, (int)(NW / 4));
  cvt_bf16_kernel<<<(int)(NW / 4 / 256), 256, 0, stream>>>(wk, wkb, (int)(NW / 4));
  cvt_bf16_kernel<<<(int)(NW / 4 / 256), 256, 0, stream>>>(wv, wvb, (int)(NW / 4));
  cvt_bf16_kernel<<<(int)(NW / 4 / 256), 256, 0, stream>>>(wo, wob, (int)(NW / 4));

  qkv_rope_kernel<<<dim3(8192 / 128, DMODEL / 128, 3), 256, 0, stream>>>(
      xb, wqb, wkb, wvb, qm, km, vt);

  flash_attn_kernel<<<dim3(S_LEN / 128, 4 * HEADS), 128, 0, stream>>>(qm, km, vt, ao);

  out_proj_kernel<<<dim3(8192 / 128, DMODEL / 128), 256, 0, stream>>>(
      ao, wob, (float*)d_out);
}
